// KNNGNN_method_16999480557902
// MI455X (gfx1250) — matrix operand order repaired and checked by reading, not compiled, not run on hardware
//
#include <hip/hip_runtime.h>
#include <hip/hip_bf16.h>

// ---------------- problem constants (from reference) ----------------
#define NN    10000
#define F_IN  256
#define HID   128
#define PROJ  64
#define KNN   50
#define HEADS 8

// ---------------- CDNA5 WMMA types ----------------
typedef __attribute__((ext_vector_type(16))) __bf16 v16bf;
typedef __attribute__((ext_vector_type(8)))  __bf16 v8bf;
typedef __attribute__((ext_vector_type(8)))  float  v8f;

__device__ __forceinline__ __bf16 f2bf(float f) {
  union { float f; unsigned u; } v; v.f = f;
  unsigned r = (v.u + 0x7FFFu + ((v.u >> 16) & 1u)) >> 16;
  unsigned short s = (unsigned short)r;
  return __builtin_bit_cast(__bf16, s);
}

// monotone float<->uint order mapping for atomicMax on floats
__device__ __forceinline__ unsigned fenc(float f) {
  unsigned b = __float_as_uint(f);
  return (b & 0x80000000u) ? ~b : (b | 0x80000000u);
}
__device__ __forceinline__ float fdec(unsigned u) {
  unsigned b = (u & 0x80000000u) ? (u ^ 0x80000000u) : ~u;
  return __uint_as_float(b);
}

// ---------------- fragment loaders (ISA 7.12.2 layouts) ----------------
// A: 16x32 bf16 row-major, lane l covers row m=(l&15); lane<16 -> K {0..7,16..23},
// lane>=16 -> K {8..15,24..31}. Two contiguous 16B loads.
__device__ __forceinline__ v16bf load_a_frag(const __bf16* A, int lda, int m0, int kt, int lane) {
  int m  = m0 + (lane & 15);
  int kb = (lane & 16) ? 8 : 0;
  const __bf16* p = A + (size_t)m * lda + kt * 32 + kb;
  v8bf lo = *(const v8bf*)(p);
  v8bf hi = *(const v8bf*)(p + 16);
  return __builtin_shufflevector(lo, hi, 0,1,2,3,4,5,6,7,8,9,10,11,12,13,14,15);
}

// same, with row clamping (for the ragged last stripe of the Gram kernel)
__device__ __forceinline__ v16bf load_a_frag_cl(const __bf16* A, int lda, int m0, int kt, int lane) {
  int m  = m0 + (lane & 15);
  m = m < NN ? m : NN - 1;
  int kb = (lane & 16) ? 8 : 0;
  const __bf16* p = A + (size_t)m * lda + kt * 32 + kb;
  v8bf lo = *(const v8bf*)(p);
  v8bf hi = *(const v8bf*)(p + 16);
  return __builtin_shufflevector(lo, hi, 0,1,2,3,4,5,6,7,8,9,10,11,12,13,14,15);
}

// B = W^T with W row-major [N,K]: B[k][n] = W[n][k]; lane l covers col n=(l&15),
// lane<16 -> K {0..7,16..23}, lane>=16 -> K {8..15,24..31}. Two contiguous 16B loads.
__device__ __forceinline__ v16bf load_b_frag(const __bf16* W, int ldw, int n0, int kt, int lane) {
  int n  = n0 + (lane & 15);
  int kb = (lane & 16) ? 8 : 0;
  const __bf16* p = W + (size_t)n * ldw + kt * 32 + kb;
  v8bf lo = *(const v8bf*)(p);
  v8bf hi = *(const v8bf*)(p + 16);
  return __builtin_shufflevector(lo, hi, 0,1,2,3,4,5,6,7,8,9,10,11,12,13,14,15);
}

// ------- generic bf16 WMMA GEMM: C = A[M,K] * W[N,K]^T (+bias)(+relu) -------
// one wave per 16x64 tile (4 accumulators share one A fragment -> 4x A reuse,
// 4 independent back-to-back WMMAs per K-step). M mult of 16, N mult of 64,
// K mult of 32.
template<bool RELU>
__global__ void k_wmma_gemm(const __bf16* __restrict__ A, int lda,
                            const __bf16* __restrict__ W, int ldw,
                            const float* __restrict__ bias,
                            float* __restrict__ Cf, int ldc,
                            __bf16* __restrict__ Cbf, int ldbf,
                            int K) {
  int lane = threadIdx.x;
  int n0 = blockIdx.x * 64;
  int m0 = blockIdx.y * 16;
  v8f acc[4];
#pragma unroll
  for (int t = 0; t < 4; ++t) acc[t] = (v8f){};
  int kt_n = K >> 5;
  for (int kt = 0; kt < kt_n; ++kt) {
    v16bf a = load_a_frag(A, lda, m0, kt, lane);
#pragma unroll
    for (int t = 0; t < 4; ++t) {
      v16bf b = load_b_frag(W, ldw, n0 + 16 * t, kt, lane);
      acc[t] = __builtin_amdgcn_wmma_f32_16x16x32_bf16(false, a, false, b, (short)0, acc[t], false, false);
    }
  }
  // D layout: lane l owns col n=(l&15); VGPR v -> row m = v + 8*(l>=16)
  int mh = m0 + ((lane & 16) ? 8 : 0);
#pragma unroll
  for (int t = 0; t < 4; ++t) {
    int nl = n0 + 16 * t + (lane & 15);
    float bv = bias ? bias[nl] : 0.0f;
#pragma unroll
    for (int v = 0; v < 8; ++v) {
      float r = acc[t][v] + bv;
      if (RELU) r = fmaxf(r, 0.0f);
      int m = mh + v;
      if (Cf)  Cf[(size_t)m * ldc + nl] = r;
      if (Cbf) Cbf[(size_t)m * ldbf + nl] = f2bf(r);
    }
  }
}

// ---------------- misc elementwise kernels ----------------
__global__ void k_f32_to_bf16(const float* __restrict__ in, __bf16* __restrict__ out, int n) {
  int t = blockIdx.x * blockDim.x + threadIdx.x;
  if (t < n) out[t] = f2bf(in[t]);
}

// attention logits: al[n,h] = sum_c h[n,h*C+c] * a[h,c]
__global__ void k_att_logits(const float* __restrict__ h, const float* __restrict__ a_src,
                             const float* __restrict__ a_dst,
                             float* __restrict__ al_src, float* __restrict__ al_dst,
                             int H, int C) {
  int t = blockIdx.x * blockDim.x + threadIdx.x;
  if (t >= NN * H) return;
  int n = t / H, hh = t - n * H;
  const float* hp = h + (size_t)n * H * C + hh * C;
  float ss = 0.f, sd = 0.f;
  for (int c = 0; c < C; ++c) {
    float x = hp[c];
    ss += x * a_src[hh * C + c];
    sd += x * a_dst[hh * C + c];
  }
  al_src[t] = ss; al_dst[t] = sd;
}

__device__ __forceinline__ float edge_logit(const float* al_src, const float* al_dst,
                                            int s, int d, int hh, int H) {
  float v = al_src[s * H + hh] + al_dst[d * H + hh];
  return v > 0.f ? v : 0.2f * v;   // leaky_relu 0.2
}

__global__ void k_edge_max(const long long* __restrict__ src, const long long* __restrict__ dst,
                           const float* __restrict__ al_src, const float* __restrict__ al_dst,
                           unsigned* __restrict__ emax, int E, int H) {
  int t = blockIdx.x * blockDim.x + threadIdx.x;
  if (t >= E * H) return;
  int e = t / H, hh = t - e * H;
  int s = (int)src[e], d = (int)dst[e];
  float v = edge_logit(al_src, al_dst, s, d, hh, H);
  atomicMax(&emax[d * H + hh], fenc(v));
}

__global__ void k_edge_denom(const long long* __restrict__ src, const long long* __restrict__ dst,
                             const float* __restrict__ al_src, const float* __restrict__ al_dst,
                             const unsigned* __restrict__ emax, float* __restrict__ denom,
                             int E, int H) {
  int t = blockIdx.x * blockDim.x + threadIdx.x;
  if (t >= E * H) return;
  int e = t / H, hh = t - e * H;
  int s = (int)src[e], d = (int)dst[e];
  float v = edge_logit(al_src, al_dst, s, d, hh, H);
  float ex = __expf(v - fdec(emax[d * H + hh]));
  atomicAdd(&denom[d * H + hh], ex);
}

// one wave per (edge, head); lanes stride the C channels of the scatter-add
__global__ void k_edge_aggregate(const long long* __restrict__ src, const long long* __restrict__ dst,
                                 const float* __restrict__ al_src, const float* __restrict__ al_dst,
                                 const unsigned* __restrict__ emax, const float* __restrict__ denom,
                                 const float* __restrict__ h, float* __restrict__ agg,
                                 int E, int H, int C) {
  int gid  = blockIdx.x * blockDim.x + threadIdx.x;
  int pair = gid >> 5;
  int lane = gid & 31;
  if (pair >= E * H) return;
  int e = pair / H, hh = pair - e * H;
  int s = (int)src[e], d = (int)dst[e];
  float v  = edge_logit(al_src, al_dst, s, d, hh, H);
  float ex = __expf(v - fdec(emax[d * H + hh]));
  float alpha = ex / (denom[d * H + hh] + 1e-16f);
  const float* hs = h   + (size_t)s * H * C + hh * C;
  float*       ad = agg + (size_t)d * H * C + hh * C;
  for (int c = lane; c < C; c += 32)
    atomicAdd(&ad[c], hs[c] * alpha);
}

// xg = relu(agg1 + b) -> bf16
__global__ void k_gat1_epilogue(const float* __restrict__ agg, const float* __restrict__ b,
                                __bf16* __restrict__ xgbf, int total) {
  int t = blockIdx.x * blockDim.x + threadIdx.x;
  if (t >= total) return;
  int c = t & 1023;
  xgbf[t] = f2bf(fmaxf(agg[t] + b[c], 0.0f));
}

// x2 = agg2 + b2 -> final[:,128:256] (f32) and concatbf[:,128:256] (bf16)
__global__ void k_gat2_epilogue(const float* __restrict__ agg2, const float* __restrict__ b2,
                                float* __restrict__ finalv, __bf16* __restrict__ concatbf) {
  int t = blockIdx.x * blockDim.x + threadIdx.x;
  if (t >= NN * HID) return;
  int n = t >> 7, c = t & 127;
  float v = agg2[t] + b2[c];
  finalv[(size_t)n * 256 + 128 + c] = v;
  concatbf[(size_t)n * 512 + 128 + c] = f2bf(v);
}

__global__ void k_sqnorm(const float* __restrict__ g, float* __restrict__ sq) {
  int n = blockIdx.x * blockDim.x + threadIdx.x;
  if (n >= NN) return;
  const float* p = g + (size_t)n * PROJ;
  float s = 0.f;
  for (int c = 0; c < PROJ; ++c) s += p[c] * p[c];
  sq[n] = s;
}

// ---------------- fused Gram + top-K (one wave per 32-row stripe) ----------------
__device__ __forceinline__ void heap_insert(float* hv, int* hi, int& cnt, float val, int id) {
  if (cnt < KNN) {                       // push, sift up (min-heap on value)
    int i = cnt++;
    hv[i] = val; hi[i] = id;
    while (i > 0) {
      int p = (i - 1) >> 1;
      if (hv[p] <= hv[i]) break;
      float tv = hv[p]; hv[p] = hv[i]; hv[i] = tv;
      int   ti = hi[p]; hi[p] = hi[i]; hi[i] = ti;
      i = p;
    }
  } else if (val > hv[0]) {              // replace root, sift down
    hv[0] = val; hi[0] = id;
    int i = 0;
    for (;;) {
      int l = 2 * i + 1, r = l + 1, s = i;
      if (l < KNN && hv[l] < hv[s]) s = l;
      if (r < KNN && hv[r] < hv[s]) s = r;
      if (s == i) break;
      float tv = hv[s]; hv[s] = hv[i]; hv[i] = tv;
      int   ti = hi[s]; hi[s] = hi[i]; hi[i] = ti;
      i = s;
    }
  }
}

// 32 rows per wave: 4 WMMAs per B-tile load, every lane owns one row heap.
__global__ void k_gram_topk(const __bf16* __restrict__ G, const float* __restrict__ sq,
                            int* __restrict__ topk) {
  __shared__ float tile[32][17];
  __shared__ float hval[32][KNN];
  __shared__ int   hidx[32][KNN];
  int lane = threadIdx.x;
  int m0 = blockIdx.x * 32;
  // preload A fragments for rows m0..m0+15 and m0+16..m0+31 (K = 64 -> 2 chunks),
  // with row clamping for the ragged last stripe (phantom rows never written out)
  v16bf a00 = load_a_frag_cl(G, PROJ, m0,      0, lane);
  v16bf a01 = load_a_frag_cl(G, PROJ, m0,      1, lane);
  v16bf a10 = load_a_frag_cl(G, PROJ, m0 + 16, 0, lane);
  v16bf a11 = load_a_frag_cl(G, PROJ, m0 + 16, 1, lane);
  float sqm0[8], sqm1[8];
  int mh = (lane & 16) ? 8 : 0;
#pragma unroll
  for (int v = 0; v < 8; ++v) {
    int r0 = m0 + mh + v,      r1 = m0 + 16 + mh + v;
    sqm0[v] = sq[r0 < NN ? r0 : NN - 1];
    sqm1[v] = sq[r1 < NN ? r1 : NN - 1];
  }
  int cnt = 0;
  const int ntiles = NN / 16;
  for (int j = 0; j < ntiles; ++j) {
    int j0 = j * 16;
    if (j + 1 < ntiles)    // warm L2/WGP$ for next column tile
      __builtin_prefetch(G + (size_t)(j0 + 16 + (lane & 15)) * PROJ, 0, 3);
    v16bf b0 = load_b_frag(G, PROJ, j0, 0, lane);   // B = G^T: same row-major trick
    v16bf b1 = load_b_frag(G, PROJ, j0, 1, lane);
    v8f acc0 = {}, acc1 = {};
    acc0 = __builtin_amdgcn_wmma_f32_16x16x32_bf16(false, a00, false, b0, (short)0, acc0, false, false);
    acc1 = __builtin_amdgcn_wmma_f32_16x16x32_bf16(false, a10, false, b0, (short)0, acc1, false, false);
    acc0 = __builtin_amdgcn_wmma_f32_16x16x32_bf16(false, a01, false, b1, (short)0, acc0, false, false);
    acc1 = __builtin_amdgcn_wmma_f32_16x16x32_bf16(false, a11, false, b1, (short)0, acc1, false, false);
    int nl = lane & 15;
    float sqj = sq[j0 + nl];
#pragma unroll
    for (int v = 0; v < 8; ++v) {
      tile[mh + v][nl]      = 2.0f * acc0[v] - sqm0[v] - sqj;   // neg_d2, rows 0..15
      tile[16 + mh + v][nl] = 2.0f * acc1[v] - sqm1[v] - sqj;   // neg_d2, rows 16..31
    }
    __syncthreads();
    {
      float* hv = hval[lane];    // every lane owns one row
      int*   hi = hidx[lane];
      for (int c = 0; c < 16; ++c)
        heap_insert(hv, hi, cnt, tile[lane][c], j0 + c);
    }
    __syncthreads();
  }
  int row = m0 + lane;
  if (row < NN) {
    int* out = topk + (size_t)row * KNN;
    for (int t = 0; t < KNN; ++t) out[t] = hidx[lane][t];
  }
}

// sim[n,c] = mean_k final[idx[n,k], c]
__global__ void k_gather_mean(const int* __restrict__ topk, const float* __restrict__ finalv,
                              float* __restrict__ sim) {
  __shared__ int idx[KNN];
  int n = blockIdx.x, c = threadIdx.x;
  if (c < KNN) idx[c] = topk[(size_t)n * KNN + c];
  __syncthreads();
  float s = 0.f;
  for (int k = 0; k < KNN; ++k) s += finalv[(size_t)idx[k] * 256 + c];
  sim[(size_t)n * 256 + c] = s * (1.0f / KNN);
}

// LayerNorm(sim) * g + b -> concatbf[:, 256:512] (bf16)
__global__ void k_layernorm_bf16(const float* __restrict__ sim, const float* __restrict__ g,
                                 const float* __restrict__ b, __bf16* __restrict__ concatbf) {
  __shared__ float red[256];
  int n = blockIdx.x, c = threadIdx.x;
  float v = sim[(size_t)n * 256 + c];
  red[c] = v; __syncthreads();
  for (int s = 128; s > 0; s >>= 1) { if (c < s) red[c] += red[c + s]; __syncthreads(); }
  float mu = red[0] * (1.0f / 256.0f);
  __syncthreads();
  float d = v - mu;
  red[c] = d * d; __syncthreads();
  for (int s = 128; s > 0; s >>= 1) { if (c < s) red[c] += red[c + s]; __syncthreads(); }
  float var = red[0] * (1.0f / 256.0f);
  float y = d * rsqrtf(var + 1e-5f) * g[c] + b[c];
  concatbf[(size_t)n * 512 + 256 + c] = f2bf(y);
}

// ---------------- host orchestration ----------------
static inline dim3 cdiv1(int n, int b) { return dim3((unsigned)((n + b - 1) / b)); }

extern "C" void kernel_launch(void* const* d_in, const int* in_sizes, int n_in,
                              void* d_out, int out_size, void* d_ws, size_t ws_size,
                              hipStream_t stream) {
  const float* x        = (const float*)d_in[0];
  const long long* ei   = (const long long*)d_in[1];   // int64 per reference
  const float* lin1_w   = (const float*)d_in[2];
  const float* lin1_b   = (const float*)d_in[3];
  const float* gat1_w   = (const float*)d_in[4];
  const float* gat1_as  = (const float*)d_in[5];
  const float* gat1_ad  = (const float*)d_in[6];
  const float* gat1_b   = (const float*)d_in[7];
  const float* gat2_w   = (const float*)d_in[8];
  const float* gat2_as  = (const float*)d_in[9];
  const float* gat2_ad  = (const float*)d_in[10];
  const float* gat2_b   = (const float*)d_in[11];
  const float* proj_w   = (const float*)d_in[12];
  const float* proj_b   = (const float*)d_in[13];
  const float* ln_g     = (const float*)d_in[14];
  const float* ln_b     = (const float*)d_in[15];
  const float* lin_w    = (const float*)d_in[16];
  const float* lin_b    = (const float*)d_in[17];
  const int E = in_sizes[1] / 2;
  const long long* src = ei;
  const long long* dst = ei + E;

  // bump allocator over d_ws
  char* wp = (char*)d_ws;
  auto alloc = [&](size_t bytes) -> void* {
    void* p = (void*)wp;
    wp += (bytes + 255) & ~(size_t)255;
    return p;
  };
  __bf16* xbf      = (__bf16*)alloc((size_t)NN * F_IN * 2);
  __bf16* w_lin1   = (__bf16*)alloc((size_t)HID * F_IN * 2);
  __bf16* w_gat1   = (__bf16*)alloc((size_t)HEADS * HID * F_IN * 2);
  __bf16* w_gat2   = (__bf16*)alloc((size_t)HID * HEADS * HID * 2);
  __bf16* w_proj   = (__bf16*)alloc((size_t)PROJ * 2 * HID * 2);
  __bf16* w_lin    = (__bf16*)alloc((size_t)64 * 4 * HID * 2);
  float*  h1       = (float*)alloc((size_t)NN * HEADS * HID * 4);
  float*  agg1     = (float*)alloc((size_t)NN * HEADS * HID * 4);
  float*  al1s     = (float*)alloc((size_t)NN * HEADS * 4);
  float*  al1d     = (float*)alloc((size_t)NN * HEADS * 4);
  unsigned* emax1  = (unsigned*)alloc((size_t)NN * HEADS * 4);
  float*  den1     = (float*)alloc((size_t)NN * HEADS * 4);
  __bf16* xgbf     = (__bf16*)alloc((size_t)NN * HEADS * HID * 2);
  float*  h2       = (float*)alloc((size_t)NN * HID * 4);
  float*  agg2     = (float*)alloc((size_t)NN * HID * 4);
  float*  al2s     = (float*)alloc((size_t)NN * 4);
  float*  al2d     = (float*)alloc((size_t)NN * 4);
  unsigned* emax2  = (unsigned*)alloc((size_t)NN * 4);
  float*  den2     = (float*)alloc((size_t)NN * 4);
  float*  finalv   = (float*)alloc((size_t)NN * 256 * 4);
  __bf16* concatbf = (__bf16*)alloc((size_t)NN * 512 * 2);
  float*  gsc      = (float*)alloc((size_t)NN * PROJ * 4);
  __bf16* gbf      = (__bf16*)alloc((size_t)NN * PROJ * 2);
  float*  sqn      = (float*)alloc((size_t)NN * 4);
  int*    topk     = (int*)alloc((size_t)NN * KNN * 4);
  float*  sim      = (float*)alloc((size_t)NN * 256 * 4);
  (void)ws_size; (void)n_in; (void)out_size;

  const int B = 256;
  // --- bf16 conversions (activations + weights) ---
  k_f32_to_bf16<<<cdiv1(NN * F_IN, B), B, 0, stream>>>(x, xbf, NN * F_IN);
  k_f32_to_bf16<<<cdiv1(HID * F_IN, B), B, 0, stream>>>(lin1_w, w_lin1, HID * F_IN);
  k_f32_to_bf16<<<cdiv1(HEADS * HID * F_IN, B), B, 0, stream>>>(gat1_w, w_gat1, HEADS * HID * F_IN);
  k_f32_to_bf16<<<cdiv1(HID * HEADS * HID, B), B, 0, stream>>>(gat2_w, w_gat2, HID * HEADS * HID);
  k_f32_to_bf16<<<cdiv1(PROJ * 2 * HID, B), B, 0, stream>>>(proj_w, w_proj, PROJ * 2 * HID);
  k_f32_to_bf16<<<cdiv1(64 * 4 * HID, B), B, 0, stream>>>(lin_w, w_lin, 64 * 4 * HID);

  // --- G1: x1 = relu(x @ lin1_w^T + b) -> final[:, :128] + concatbf[:, :128] ---
  k_wmma_gemm<true><<<dim3(HID / 64, NN / 16), 32, 0, stream>>>(
      xbf, F_IN, w_lin1, F_IN, lin1_b, finalv, 256, concatbf, 512, F_IN);

  // --- G2: h1 = x @ gat1_w^T (no bias) ---
  k_wmma_gemm<false><<<dim3(HEADS * HID / 64, NN / 16), 32, 0, stream>>>(
      xbf, F_IN, w_gat1, F_IN, nullptr, h1, HEADS * HID, (__bf16*)nullptr, 0, F_IN);

  // --- GAT layer 1 (H=8, C=128) ---
  k_att_logits<<<cdiv1(NN * HEADS, B), B, 0, stream>>>(h1, gat1_as, gat1_ad, al1s, al1d, HEADS, HID);
  hipMemsetAsync(emax1, 0, (size_t)NN * HEADS * 4, stream);
  hipMemsetAsync(den1,  0, (size_t)NN * HEADS * 4, stream);
  hipMemsetAsync(agg1,  0, (size_t)NN * HEADS * HID * 4, stream);
  k_edge_max<<<cdiv1(E * HEADS, B), B, 0, stream>>>(src, dst, al1s, al1d, emax1, E, HEADS);
  k_edge_denom<<<cdiv1(E * HEADS, B), B, 0, stream>>>(src, dst, al1s, al1d, emax1, den1, E, HEADS);
  k_edge_aggregate<<<cdiv1(E * HEADS * 32, B), B, 0, stream>>>(
      src, dst, al1s, al1d, emax1, den1, h1, agg1, E, HEADS, HID);
  k_gat1_epilogue<<<cdiv1(NN * HEADS * HID, B), B, 0, stream>>>(agg1, gat1_b, xgbf, NN * HEADS * HID);

  // --- G3: h2 = xg @ gat2_w^T (no bias) ---
  k_wmma_gemm<false><<<dim3(HID / 64, NN / 16), 32, 0, stream>>>(
      xgbf, HEADS * HID, w_gat2, HEADS * HID, nullptr, h2, HID, (__bf16*)nullptr, 0, HEADS * HID);

  // --- GAT layer 2 (H=1, C=128) ---
  k_att_logits<<<cdiv1(NN, B), B, 0, stream>>>(h2, gat2_as, gat2_ad, al2s, al2d, 1, HID);
  hipMemsetAsync(emax2, 0, (size_t)NN * 4, stream);
  hipMemsetAsync(den2,  0, (size_t)NN * 4, stream);
  hipMemsetAsync(agg2,  0, (size_t)NN * HID * 4, stream);
  k_edge_max<<<cdiv1(E, B), B, 0, stream>>>(src, dst, al2s, al2d, emax2, E, 1);
  k_edge_denom<<<cdiv1(E, B), B, 0, stream>>>(src, dst, al2s, al2d, emax2, den2, E, 1);
  k_edge_aggregate<<<cdiv1(E * 32, B), B, 0, stream>>>(
      src, dst, al2s, al2d, emax2, den2, h2, agg2, E, 1, HID);
  k_gat2_epilogue<<<cdiv1(NN * HID, B), B, 0, stream>>>(agg2, gat2_b, finalv, concatbf);

  // --- G4: g_score = final @ proj_w^T + b  (A = concatbf[:, :256], lda=512) ---
  k_wmma_gemm<false><<<dim3(PROJ / 64, NN / 16), 32, 0, stream>>>(
      concatbf, 512, w_proj, 2 * HID, proj_b, gsc, PROJ, gbf, PROJ, 2 * HID);

  // --- fused Gram + top-K ---
  k_sqnorm<<<cdiv1(NN, B), B, 0, stream>>>(gsc, sqn);
  k_gram_topk<<<cdiv1(NN, 32), 32, 0, stream>>>(gbf, sqn, topk);

  // --- gather-mean + LayerNorm -> concatbf[:, 256:512] ---
  k_gather_mean<<<NN, 256, 0, stream>>>(topk, finalv, sim);
  k_layernorm_bf16<<<NN, 256, 0, stream>>>(sim, ln_g, ln_b, concatbf);

  // --- G6: out = concat(final, sim_ln) @ lin_w^T + b ---
  k_wmma_gemm<false><<<dim3(64 / 64, NN / 16), 32, 0, stream>>>(
      concatbf, 512, w_lin, 4 * HID, lin_b, (float*)d_out, 64, (__bf16*)nullptr, 0, 4 * HID);
}
